// RecurrentLanguageModel_33406255629241
// MI455X (gfx1250) — matrix operand order repaired and checked by reading, not compiled, not run on hardware
//
#include <hip/hip_runtime.h>
#include <hip/hip_bf16.h>
#include <math.h>

typedef __bf16 bf16_t;
typedef __attribute__((ext_vector_type(16))) __bf16 v16bf;
typedef __attribute__((ext_vector_type(8)))  __bf16 v8bf;
typedef __attribute__((ext_vector_type(8)))  float  v8f;

#define BATCH 8
#define SEQ   512
#define DIM   1024
#define VOCAB 32064
#define NLAYER 2
#define RNN_BLOCKS 16

static __device__ __forceinline__ v8f zero_v8f() {
  v8f z = {0.f, 0.f, 0.f, 0.f, 0.f, 0.f, 0.f, 0.f};
  return z;
}

// A fragment (16x32 bf16, M x K) from row-major [16][kstride] buffer.
// lane<16 : row=lane,    K = {k0..k0+15}
// lane>=16: row=lane-16, K = {k0+16..k0+31}
static __device__ __forceinline__ v16bf load_a_frag(const bf16_t* __restrict__ base,
                                                    int kstride, int row, int sel, int k0) {
  const bf16_t* p = base + row * kstride + k0 + sel * 16;
  v8bf a0 = *(const v8bf*)(p);
  v8bf a1 = *(const v8bf*)(p + 8);
  return __builtin_shufflevector(a0, a1, 0, 1, 2, 3, 4, 5, 6, 7,
                                 8, 9, 10, 11, 12, 13, 14, 15);
}

// Same fragment layout, reading from LDS.
static __device__ __forceinline__ v16bf load_a_frag_lds(const bf16_t* base,
                                                        int row, int sel, int k0) {
  const bf16_t* p = base + row * DIM + k0 + sel * 16;
  v8bf a0 = *(const v8bf*)(p);
  v8bf a1 = *(const v8bf*)(p + 8);
  return __builtin_shufflevector(a0, a1, 0, 1, 2, 3, 4, 5, 6, 7,
                                 8, 9, 10, 11, 12, 13, 14, 15);
}

// ---------------------------------------------------------------------------
// Async-stage 32KB (16 rows x 1024 bf16) global -> LDS with the CDNA5
// GLOBAL_LOAD_ASYNC_TO_LDS path (tracked by ASYNCcnt, no VGPR data movement).
// 128 threads, 16 chunks of 16B each per thread.
// ---------------------------------------------------------------------------
static __device__ __forceinline__ void stage32k_async(const bf16_t* __restrict__ g,
                                                      bf16_t* l, int tid) {
  unsigned lbase = (unsigned)(unsigned long long)l;  // low 32 bits = LDS offset
#pragma unroll
  for (int i = 0; i < 16; ++i) {
    int idx = i * 128 + tid;            // 16-byte chunk index, 2048 total
    unsigned laddr = lbase + (unsigned)idx * 16u;
    unsigned goff = (unsigned)idx * 16u;
    asm volatile("global_load_async_to_lds_b128 %0, %1, %2 offset:0"
                 :: "v"(laddr), "v"(goff), "s"(g)
                 : "memory");
  }
}

static __device__ __forceinline__ void wait_async_all() {
  asm volatile("s_wait_asynccnt 0x0" ::: "memory");
}

// Device-wide barrier for a small co-resident grid (monotonic counter).
static __device__ __forceinline__ void grid_sync(unsigned* ctr, unsigned target) {
  __syncthreads();
  if (threadIdx.x == 0) {
    __hip_atomic_fetch_add(ctr, 1u, __ATOMIC_RELEASE, __HIP_MEMORY_SCOPE_AGENT);
    while (__hip_atomic_load(ctr, __ATOMIC_ACQUIRE, __HIP_MEMORY_SCOPE_AGENT) < target) {
      __builtin_amdgcn_s_sleep(2);
    }
  }
  __syncthreads();
}

// ---------------------------------------------------------------------------
// Prep: Wh/Uh -> combined transposed bf16 weights wt[l][n][k], k in [0,2048)
// ---------------------------------------------------------------------------
__global__ void rlm_cvt_wu_kernel(const float* __restrict__ Wh,
                                  const float* __restrict__ Uh,
                                  bf16_t* __restrict__ wt) {
  int idx = blockIdx.x * blockDim.x + threadIdx.x;
  if (idx >= NLAYER * DIM * 2 * DIM) return;
  int k = idx & 2047;
  int n = (idx >> 11) & 1023;
  int l = idx >> 21;
  float v = (k < DIM) ? Wh[((size_t)l * DIM + k) * DIM + n]
                      : Uh[((size_t)l * DIM + (k - DIM)) * DIM + n];
  wt[idx] = (bf16_t)v;
}

// ---------------------------------------------------------------------------
// Prep: fc_w [DIM][VOCAB] f32 -> fcwt [VOCAB][DIM] bf16 (LDS-tiled transpose)
// ---------------------------------------------------------------------------
__global__ void rlm_cvt_fcw_kernel(const float* __restrict__ fcw,
                                   bf16_t* __restrict__ fcwt) {
  __shared__ float tile[32][33];
  int nbase = blockIdx.x * 32;
  int kbase = blockIdx.y * 32;
  int tx = threadIdx.x;
  int ty = threadIdx.y;
#pragma unroll
  for (int j = 0; j < 4; ++j)
    tile[ty + j * 8][tx] = fcw[(size_t)(kbase + ty + j * 8) * VOCAB + nbase + tx];
  __syncthreads();
#pragma unroll
  for (int j = 0; j < 4; ++j)
    fcwt[(size_t)(nbase + ty + j * 8) * DIM + kbase + tx] =
        (bf16_t)tile[tx][ty + j * 8];
}

// ---------------------------------------------------------------------------
// Prep: embedding gather -> x[t][row][d] bf16, rows 8..15 zero, pad_idx zeroed
// ---------------------------------------------------------------------------
__global__ void rlm_embed_kernel(const int* __restrict__ tokens,
                                 const float* __restrict__ emb,
                                 bf16_t* __restrict__ x) {
  int idx = blockIdx.x * blockDim.x + threadIdx.x;
  if (idx >= SEQ * 16 * DIM) return;
  int d = idx & 1023;
  int r = (idx >> 10) & 15;
  int t = idx >> 14;
  float v = 0.f;
  if (r < BATCH) {
    int tok = tokens[r * SEQ + t];
    if (tok != 0) v = emb[(size_t)tok * DIM + d];
  }
  x[idx] = (bf16_t)v;
}

__global__ void rlm_zero_bf16_kernel(bf16_t* __restrict__ p, int n) {
  int i = blockIdx.x * blockDim.x + threadIdx.x;
  if (i < n) p[i] = (bf16_t)0.f;
}

// ---------------------------------------------------------------------------
// One RNN layer for one 16-column tile: acc = [lin | lh] @ Wcomb, then
// tanh(acc + bias) -> hnew (and outh for layer 1). A from LDS, B from global.
// 4 independent WMMA accumulator chains (16 deep each) so a single wave per
// SIMD can keep the matrix pipe busy; reduced with 3 vector adds at the end.
// ---------------------------------------------------------------------------
static __device__ __forceinline__ void rnn_layer_tile(const bf16_t* lin, const bf16_t* lh,
                                                      const bf16_t* __restrict__ wt_layer,
                                                      const float* __restrict__ bias,
                                                      bf16_t* __restrict__ hnew,
                                                      bf16_t* __restrict__ outh, int t,
                                                      int nbase, int lane, int row, int sel) {
  const bf16_t* btrow = wt_layer + (size_t)(nbase + row) * 2048;
  v8f acc[4];
#pragma unroll
  for (int i = 0; i < 4; ++i) acc[i] = zero_v8f();

#pragma unroll 8
  for (int kc = 0; kc < 32; ++kc) {
    int k0 = kc * 32;
    v16bf a = load_a_frag_lds(lin, row, sel, k0);
    v16bf b = *(const v16bf*)(btrow + k0 + sel * 16);
    acc[kc & 3] = __builtin_amdgcn_wmma_f32_16x16x32_bf16(
        false, a, false, b, (short)0, acc[kc & 3], false, false);
  }
#pragma unroll 8
  for (int kc = 0; kc < 32; ++kc) {
    int k0 = kc * 32;
    v16bf a = load_a_frag_lds(lh, row, sel, k0);
    v16bf b = *(const v16bf*)(btrow + DIM + k0 + sel * 16);
    acc[kc & 3] = __builtin_amdgcn_wmma_f32_16x16x32_bf16(
        false, a, false, b, (short)0, acc[kc & 3], false, false);
  }
  v8f sum = (acc[0] + acc[1]) + (acc[2] + acc[3]);

  if (lane < 16) {
    int col = nbase + lane;
    float bv = bias[col];
#pragma unroll
    for (int m = 0; m < BATCH; ++m) {
      float hv = tanhf(sum[m] + bv);
      bf16_t hb = (bf16_t)hv;
      hnew[m * DIM + col] = hb;
      if (outh) outh[((size_t)m * SEQ + t) * DIM + col] = hb;
    }
  }
}

// ---------------------------------------------------------------------------
// Persistent RNN: whole 512-step x 2-layer recurrence in ONE kernel.
// Grid 16 blocks x 128 threads (4 waves); wave -> one 16-col tile (64 tiles).
// Per layer-step: async-stage A halves to LDS, WMMA over K=2048, tanh,
// then device-wide barrier so the next layer/step sees the new hidden state.
// ---------------------------------------------------------------------------
__global__ void __launch_bounds__(128)
rlm_rnn_persistent_kernel(const bf16_t* __restrict__ x,
                          bf16_t* __restrict__ hbuf,
                          const bf16_t* __restrict__ wt,
                          const float* __restrict__ bias,
                          bf16_t* __restrict__ outh,
                          unsigned* __restrict__ ctr) {
  __shared__ bf16_t lin[16 * DIM];  // 32KB: current layer input rows
  __shared__ bf16_t lh[16 * DIM];   // 32KB: current layer previous hidden

  const int tid = threadIdx.x;
  const int lane = tid & 31;
  const int wave = tid >> 5;
  const int tile = blockIdx.x * 4 + wave;  // 0..63
  const int nbase = tile * 16;
  const int row = lane & 15;
  const int sel = (lane >> 4) & 1;
  const size_t HS = (size_t)16 * DIM;
  const size_t WSTRIDE = (size_t)DIM * 2 * DIM;

  unsigned bar = 0;
  for (int t = 0; t < SEQ; ++t) {
    int pp = t & 1, np = pp ^ 1;
    bf16_t* h0p = hbuf + (0 * 2 + pp) * HS;
    bf16_t* h0n = hbuf + (0 * 2 + np) * HS;
    bf16_t* h1p = hbuf + (1 * 2 + pp) * HS;
    bf16_t* h1n = hbuf + (1 * 2 + np) * HS;

    // ---- layer 0: h0n = tanh([x_t | h0p] @ W0comb + b0) ----
    stage32k_async(x + (size_t)t * HS, lin, tid);
    stage32k_async(h0p, lh, tid);
    wait_async_all();
    __syncthreads();
    rnn_layer_tile(lin, lh, wt, bias, h0n, nullptr, t, nbase, lane, row, sel);
    grid_sync(ctr, RNN_BLOCKS * (++bar));

    // ---- layer 1: h1n = tanh([h0n | h1p] @ W1comb + b1) ----
    stage32k_async(h0n, lin, tid);
    stage32k_async(h1p, lh, tid);
    wait_async_all();
    __syncthreads();
    rnn_layer_tile(lin, lh, wt + WSTRIDE, bias + DIM, h1n, outh, t, nbase, lane, row, sel);
    grid_sync(ctr, RNN_BLOCKS * (++bar));
  }
}

// ---------------------------------------------------------------------------
// logits[4096][VOCAB] = outh[4096][1024] @ fcwt^T + fc_b
// grid (VOCAB/64, 4096/256), block 256 (8 waves): wave -> 32 rows x 64 cols.
// 8 independent accumulator chains per wave.
// ---------------------------------------------------------------------------
__global__ void __launch_bounds__(256)
rlm_logits_gemm_kernel(const bf16_t* __restrict__ outh,
                       const bf16_t* __restrict__ fcwt,
                       const float* __restrict__ fcb,
                       float* __restrict__ logits) {
  int lane = threadIdx.x & 31;
  int wave = threadIdx.x >> 5;
  int mbase = blockIdx.y * 256 + wave * 32;
  int nblock = blockIdx.x * 64;
  int row = lane & 15;
  int sel = (lane >> 4) & 1;

  v8f acc[2][4];
#pragma unroll
  for (int i = 0; i < 2; ++i)
#pragma unroll
    for (int j = 0; j < 4; ++j) acc[i][j] = zero_v8f();

  const bf16_t* a0 = outh + (size_t)(mbase + row) * DIM;
  const bf16_t* a1 = a0 + (size_t)16 * DIM;

#pragma unroll 2
  for (int kc = 0; kc < 32; ++kc) {
    int k0 = kc * 32;
    v16bf af0 = load_a_frag(a0, DIM, 0, sel, k0);
    v16bf af1 = load_a_frag(a1, DIM, 0, sel, k0);
#pragma unroll
    for (int j = 0; j < 4; ++j) {
      v16bf bfrag = *(const v16bf*)(fcwt + (size_t)(nblock + j * 16 + row) * DIM +
                                    k0 + sel * 16);
      acc[0][j] = __builtin_amdgcn_wmma_f32_16x16x32_bf16(
          false, af0, false, bfrag, (short)0, acc[0][j], false, false);
      acc[1][j] = __builtin_amdgcn_wmma_f32_16x16x32_bf16(
          false, af1, false, bfrag, (short)0, acc[1][j], false, false);
    }
  }

  if (lane < 16) {
#pragma unroll
    for (int mt = 0; mt < 2; ++mt)
#pragma unroll
      for (int j = 0; j < 4; ++j) {
        int col = nblock + j * 16 + lane;
        float bb = fcb[col];
#pragma unroll
        for (int m = 0; m < 8; ++m) {
          size_t r = (size_t)(mbase + mt * 16 + m);
          logits[r * VOCAB + col] = acc[mt][j][m] + bb;
        }
      }
  }
}

// ---------------------------------------------------------------------------
extern "C" void kernel_launch(void* const* d_in, const int* in_sizes, int n_in,
                              void* d_out, int out_size, void* d_ws, size_t ws_size,
                              hipStream_t stream) {
  const int*   tokens = (const int*)d_in[0];
  const float* emb    = (const float*)d_in[1];
  const float* Wh     = (const float*)d_in[2];
  const float* Uh     = (const float*)d_in[3];
  const float* bvec   = (const float*)d_in[4];
  const float* fcw    = (const float*)d_in[5];
  const float* fcb    = (const float*)d_in[6];
  float* logits = (float*)d_out;

  char* ws = (char*)d_ws;
  size_t off = 0;
  auto wsalloc = [&](size_t bytes) -> char* {
    char* p = ws + off;
    off = (off + bytes + 255) & ~(size_t)255;
    return p;
  };
  bf16_t*   wt   = (bf16_t*)wsalloc((size_t)NLAYER * DIM * 2 * DIM * 2);  //  8.39 MB
  bf16_t*   fcwt = (bf16_t*)wsalloc((size_t)VOCAB * DIM * 2);             // 65.67 MB
  bf16_t*   x    = (bf16_t*)wsalloc((size_t)SEQ * 16 * DIM * 2);          // 16.78 MB
  bf16_t*   outh = (bf16_t*)wsalloc((size_t)BATCH * SEQ * DIM * 2);       //  8.39 MB
  bf16_t*   hbuf = (bf16_t*)wsalloc((size_t)NLAYER * 2 * 16 * DIM * 2);   //  0.13 MB
  unsigned* ctr  = (unsigned*)wsalloc(256);
  (void)ws_size; (void)in_sizes; (void)n_in; (void)out_size;

  // --- prep ---
  {
    int total = NLAYER * DIM * 2 * DIM;
    rlm_cvt_wu_kernel<<<(total + 255) / 256, 256, 0, stream>>>(Wh, Uh, wt);
  }
  rlm_cvt_fcw_kernel<<<dim3(VOCAB / 32, DIM / 32), dim3(32, 8), 0, stream>>>(fcw, fcwt);
  {
    int total = SEQ * 16 * DIM;
    rlm_embed_kernel<<<(total + 255) / 256, 256, 0, stream>>>(tokens, emb, x);
  }
  {
    int total = NLAYER * 2 * 16 * DIM;
    rlm_zero_bf16_kernel<<<(total + 255) / 256, 256, 0, stream>>>(hbuf, total);
  }
  hipMemsetAsync(ctr, 0, sizeof(unsigned), stream);

  // --- whole recurrence in one persistent kernel ---
  rlm_rnn_persistent_kernel<<<RNN_BLOCKS, 128, 0, stream>>>(x, hbuf, wt, bvec, outh, ctr);

  // --- output projection ---
  rlm_logits_gemm_kernel<<<dim3(VOCAB / 64, (BATCH * SEQ) / 256), 256, 0, stream>>>(
      outh, fcwt, fcb, logits);
}
